// stream_density_wfa_7739531067865
// MI455X (gfx1250) — matrix operand order repaired and checked by reading, not compiled, not run on hardware
//
#include <hip/hip_runtime.h>
#include <hip/hip_bf16.h>
#include <cstdint>
#include <cstddef>

// Problem constants (from reference)
#define XD   128
#define DDIM 512
#define RDIM 512
#define NMIX 8
#define WIN  64
#define LOG2PI 1.8378770664093453f

typedef __attribute__((ext_vector_type(16))) __bf16    v16bf;
typedef __attribute__((ext_vector_type(8)))  float     v8f;
typedef __attribute__((ext_vector_type(8)))  unsigned  v8u;

static __device__ __forceinline__ unsigned short f2bf(float f) {
  unsigned u = __builtin_bit_cast(unsigned, f);
  u += 0x7FFFu + ((u >> 16) & 1u);          // round-to-nearest-even
  return (unsigned short)(u >> 16);
}

// ---------------------------------------------------------------------------
// Kernel 1a: H1 = tanh(X[:64] @ W1^T + b1)   [64,512] f32
// ---------------------------------------------------------------------------
__global__ void k_enc1(const float* __restrict__ X, const float* __restrict__ W1,
                       const float* __restrict__ b1, float* __restrict__ H1) {
  __shared__ float xs[XD];
  const int t = blockIdx.x;
  if (threadIdx.x < XD) xs[threadIdx.x] = X[t * XD + threadIdx.x];
  __syncthreads();
  for (int d = threadIdx.x; d < DDIM; d += blockDim.x) {
    const float* w = W1 + (size_t)d * XD;
    float s = b1[d];
    #pragma unroll 4
    for (int x = 0; x < XD; ++x) s = fmaf(xs[x], w[x], s);
    H1[t * DDIM + d] = tanhf(s);
  }
}

// ---------------------------------------------------------------------------
// Kernel 1b: E = tanh(H1 @ W2^T + b2), stored bf16 row-major [64,512]
// ---------------------------------------------------------------------------
__global__ void k_enc2(const float* __restrict__ H1, const float* __restrict__ W2,
                       const float* __restrict__ b2, unsigned short* __restrict__ Ebf) {
  __shared__ float hsrow[DDIM];
  const int t = blockIdx.x;
  for (int k = threadIdx.x; k < DDIM; k += blockDim.x) hsrow[k] = H1[t * DDIM + k];
  __syncthreads();
  for (int d = threadIdx.x; d < DDIM; d += blockDim.x) {
    const float* w = W2 + (size_t)d * DDIM;
    float s = b2[d];
    #pragma unroll 4
    for (int k = 0; k < DDIM; ++k) s = fmaf(hsrow[k], w[k], s);
    Ebf[t * DDIM + d] = f2bf(tanhf(s));
  }
}

// ---------------------------------------------------------------------------
// Kernel 2 (dominant): B[t,i,j] = sum_d E[t,d] * A[i,d,j]   (bf16 WMMA, f32 acc)
// Grid: (512 i-slices, 8 j-blocks of 64), block = 128 threads = 4 waves.
// Each wave owns a 16-wide j strip and computes a [64 x 16] output via
// 16 K-chunks x 4 M-tiles of v_wmma_f32_16x16x32_bf16. A is streamed from HBM
// exactly once (512 MB -> ~22us roofline floor @ 23.3 TB/s).
// ---------------------------------------------------------------------------
__global__ void __launch_bounds__(128)
k_core(const float* __restrict__ A, const unsigned short* __restrict__ Ebf,
       unsigned short* __restrict__ Bout) {
  const int i    = blockIdx.x;           // 0..511  (A slice)
  const int jb   = blockIdx.y;           // 0..7    (64-wide j block)
  const int wv   = threadIdx.x >> 5;     // wave in block, 0..3
  const int lane = threadIdx.x & 31;
  const int lh   = lane >> 4;            // lane-half: 0 or 1
  const int n    = lane & 15;
  const int j    = jb * 64 + wv * 16 + n;

  const float* __restrict__ Ai = A + (size_t)i * (DDIM * RDIM);

  v8f acc[4];
  #pragma unroll
  for (int mt = 0; mt < 4; ++mt) acc[mt] = (v8f)(0.0f);

  for (int kb = 0; kb < DDIM; kb += 32) {
    // ---- B operand: A_i tile [32 x 16]; lane covers K = kb + lh*16 + (0..15),
    //      column j. 16 coalesced b32 loads with immediate offsets + cvt to bf16.
    const float* __restrict__ ap = Ai + (size_t)(kb + lh * 16) * RDIM + j;
    v8u bu;
    #pragma unroll
    for (int p = 0; p < 8; ++p) {
      const float a0 = ap[(size_t)(2 * p)     * RDIM];
      const float a1 = ap[(size_t)(2 * p + 1) * RDIM];
      bu[p] = (unsigned)f2bf(a0) | ((unsigned)f2bf(a1) << 16);
    }
    const v16bf bfrag = __builtin_bit_cast(v16bf, bu);

    // ---- A operands: E tiles [16 x 32] for the 4 M(t)-tiles.
    const int khalf8 = lh * 16;
    #pragma unroll
    for (int mt = 0; mt < 4; ++mt) {
      const unsigned short* __restrict__ ep =
          Ebf + (size_t)(mt * 16 + n) * DDIM + kb + khalf8;
      const uint4 lo = *(const uint4*)(ep);
      const uint4 hi = *(const uint4*)(ep + 8);
      v8u au;
      au[0] = lo.x; au[1] = lo.y; au[2] = lo.z; au[3] = lo.w;
      au[4] = hi.x; au[5] = hi.y; au[6] = hi.z; au[7] = hi.w;
      const v16bf afrag = __builtin_bit_cast(v16bf, au);
      acc[mt] = __builtin_amdgcn_wmma_f32_16x16x32_bf16(
          /*neg_a=*/false, afrag, /*neg_b=*/false, bfrag,
          /*c_mod=*/(short)0, acc[mt], /*reuse_a=*/false, /*reuse_b=*/false);
    }
  }

  // ---- Store D tiles: lane row m = v + lh*8 within tile -> t index; bf16 out.
  #pragma unroll
  for (int mt = 0; mt < 4; ++mt) {
    #pragma unroll
    for (int v = 0; v < 8; ++v) {
      const int t = mt * 16 + lh * 8 + v;
      Bout[((size_t)t * RDIM + i) * RDIM + j] = f2bf(acc[mt][v]);
    }
  }
}

// ---------------------------------------------------------------------------
// Kernel 2b: transpose B[t][i][j] -> C[t][j][i] (bf16, 64x64 LDS tiles).
// Needed so the WMMA recurrence can load its B-operand (contraction over i)
// with contiguous b128 reads. ~66 MB of L2-resident traffic over 4096 blocks.
// LDS row stride 66 ushorts: rows dword-aligned, and the 8-row-stride write
// pattern maps to distinct banks (8*33 mod 64 = 8).
// ---------------------------------------------------------------------------
__global__ void __launch_bounds__(256)
k_transpose(const unsigned short* __restrict__ Bws, unsigned short* __restrict__ Ct) {
  __shared__ unsigned short tile[64][66];    // [j_local][i_local]
  const int t  = blockIdx.x;                 // 0..63
  const int jb = blockIdx.y;                 // 0..7
  const int ib = blockIdx.z;                 // 0..7
  const int tid = threadIdx.x;
  const int c = tid & 7;                     // 8-wide chunk index
  const int r = tid >> 3;                    // 0..31

  #pragma unroll
  for (int h = 0; h < 2; ++h) {
    const int rr = r + h * 32;               // i_local
    const uint4 v = *(const uint4*)(Bws +
        ((size_t)t * RDIM + ib * 64 + rr) * RDIM + jb * 64 + c * 8);
    const unsigned vv[4] = {v.x, v.y, v.z, v.w};
    #pragma unroll
    for (int k = 0; k < 4; ++k) {
      tile[c * 8 + 2 * k    ][rr] = (unsigned short)(vv[k] & 0xFFFFu);
      tile[c * 8 + 2 * k + 1][rr] = (unsigned short)(vv[k] >> 16);
    }
  }
  __syncthreads();
  #pragma unroll
  for (int h = 0; h < 2; ++h) {
    const int jj = r + h * 32;               // j_local
    uint4 o;
    o.x = *(const unsigned*)(&tile[jj][c * 8 + 0]);
    o.y = *(const unsigned*)(&tile[jj][c * 8 + 2]);
    o.z = *(const unsigned*)(&tile[jj][c * 8 + 4]);
    o.w = *(const unsigned*)(&tile[jj][c * 8 + 6]);
    *(uint4*)(Ct + ((size_t)t * RDIM + jb * 64 + jj) * RDIM + ib * 64 + c * 8) = o;
  }
}

// ---------------------------------------------------------------------------
// Kernel 3: sequential recurrence h <- sigmoid(h @ B_t) on the MATRIX pipe.
// One workgroup, 1024 threads = 32 waves; wave w owns j-tile [16w, 16w+16).
// Per step, per wave: 16 chunks of v_wmma_f32_16x16x32_bf16 where the
// A-operand is h broadcast into matrix row 0 (rows 1..15 zero), and the
// B-operand is C[t][j][i] loaded with two contiguous b128s per lane.
// Result row M=0 sits in acc[0] of lanes 0..15. Then the GMM head.
// ---------------------------------------------------------------------------
__global__ void __launch_bounds__(1024)
k_recur(const unsigned short* __restrict__ Ct, const float* __restrict__ prev,
        const float* __restrict__ X,
        const float* __restrict__ Wmu,  const float* __restrict__ bmu,
        const float* __restrict__ Wmu2, const float* __restrict__ bmu2,
        const float* __restrict__ Wsig, const float* __restrict__ bsig,
        const float* __restrict__ Wal,  const float* __restrict__ bal,
        float* __restrict__ out) {
  __shared__ float          hsv[RDIM];
  __shared__ unsigned short hbf[RDIM];
  __shared__ float g1[NMIX * XD];
  __shared__ float term[NMIX * XD];
  __shared__ float amix[NMIX], cmix[NMIX];

  const int tid  = threadIdx.x;
  const int w    = tid >> 5;     // wave 0..31 -> j-tile
  const int lane = tid & 31;
  const int m    = lane & 15;    // matrix row within tile (only m==0 carries h)
  const int lh   = lane >> 4;    // lane half -> K +0 / +16
  const int j    = w * 16 + m;

  if (tid < RDIM) { const float p = prev[tid]; hsv[tid] = p; hbf[tid] = f2bf(p); }
  __syncthreads();

  for (int t = 0; t < WIN; ++t) {
    const unsigned short* __restrict__ bt =
        Ct + (size_t)t * (RDIM * RDIM) + (size_t)j * RDIM + lh * 16;
    if (t + 1 < WIN) __builtin_prefetch(bt + (size_t)RDIM * RDIM, 0, 1);

    v8f acc = (v8f)(0.0f);
    #pragma unroll 4
    for (int kb = 0; kb < RDIM; kb += 32) {
      // A operand: h chunk in matrix row 0 only (lanes with m==0), zeros else.
      v8u au = (v8u)(0u);
      if (m == 0) {
        const uint4 alo = *(const uint4*)(&hbf[kb + lh * 16]);
        const uint4 ahi = *(const uint4*)(&hbf[kb + lh * 16 + 8]);
        au[0] = alo.x; au[1] = alo.y; au[2] = alo.z; au[3] = alo.w;
        au[4] = ahi.x; au[5] = ahi.y; au[6] = ahi.z; au[7] = ahi.w;
      }
      const v16bf afrag = __builtin_bit_cast(v16bf, au);

      // B operand: 16 consecutive i at K = kb + lh*16, column j. Two b128s.
      const uint4 q0 = *(const uint4*)(bt + kb);
      const uint4 q1 = *(const uint4*)(bt + kb + 8);
      v8u bu;
      bu[0] = q0.x; bu[1] = q0.y; bu[2] = q0.z; bu[3] = q0.w;
      bu[4] = q1.x; bu[5] = q1.y; bu[6] = q1.z; bu[7] = q1.w;
      const v16bf bfrag = __builtin_bit_cast(v16bf, bu);

      acc = __builtin_amdgcn_wmma_f32_16x16x32_bf16(
          false, afrag, false, bfrag, (short)0, acc, false, false);
    }
    __syncthreads();                   // all reads of hbf done
    if (lh == 0) {                     // lanes 0..15 hold row M=0 in acc[0]
      const float hnew = 1.0f / (1.0f + expf(-acc[0]));
      hsv[j] = hnew;
      hbf[j] = f2bf(hnew);
    }
    __syncthreads();
  }

  // ---- Head: one q per thread (q = 0..1023)
  const int q = tid;
  {
    const float* __restrict__ wv = Wmu + (size_t)q * RDIM;
    float s = bmu[q];
    #pragma unroll 4
    for (int r = 0; r < RDIM; ++r) s = fmaf(hsv[r], wv[r], s);
    g1[q] = tanhf(s);
  }
  __syncthreads();
  {
    const float* __restrict__ w2 = Wmu2 + (size_t)q * (NMIX * XD);
    float muv = bmu2[q];
    #pragma unroll 4
    for (int p = 0; p < NMIX * XD; ++p) muv = fmaf(g1[p], w2[p], muv);

    const float* __restrict__ wsp = Wsig + (size_t)q * RDIM;
    float sv = bsig[q];
    #pragma unroll 4
    for (int r = 0; r < RDIM; ++r) sv = fmaf(hsv[r], wsp[r], sv);

    const float nxt  = X[WIN * XD + (q & (XD - 1))];
    const float isig = expf(-sv);      // 1/sigma
    const float z    = (nxt - muv) * isig;
    term[q] = z * z + 2.0f * sv + LOG2PI;
  }
  __syncthreads();

  if (tid < NMIX) {
    float ssum = 0.0f;
    for (int x = 0; x < XD; ++x) ssum += term[tid * XD + x];
    cmix[tid] = -0.5f * ssum;
    const float* __restrict__ wa = Wal + (size_t)tid * RDIM;
    float av = bal[tid];
    for (int r = 0; r < RDIM; ++r) av = fmaf(hsv[r], wa[r], av);
    amix[tid] = av;
  }
  __syncthreads();

  if (tid == 0) {
    // ll = lse(a + comp) - lse(a)
    float m1 = -3.4e38f, m2 = -3.4e38f;
    for (int mm = 0; mm < NMIX; ++mm) {
      m1 = fmaxf(m1, amix[mm] + cmix[mm]);
      m2 = fmaxf(m2, amix[mm]);
    }
    float s1 = 0.0f, s2 = 0.0f;
    for (int mm = 0; mm < NMIX; ++mm) {
      s1 += expf(amix[mm] + cmix[mm] - m1);
      s2 += expf(amix[mm] - m2);
    }
    out[0] = (m1 + logf(s1)) - (m2 + logf(s2));
  }
}

// ---------------------------------------------------------------------------
// Launch: enc1 -> enc2 -> WMMA core -> transpose -> WMMA recurrence + head.
// Workspace: H1 f32 (128 KB) | E bf16 (64 KB) | B bf16 (32 MB) | C bf16 (32 MB)
//            ~= 64.2 MB total.
// ---------------------------------------------------------------------------
extern "C" void kernel_launch(void* const* d_in, const int* in_sizes, int n_in,
                              void* d_out, int out_size, void* d_ws, size_t ws_size,
                              hipStream_t stream) {
  (void)in_sizes; (void)n_in; (void)out_size; (void)ws_size;
  const float* prev = (const float*)d_in[0];
  const float* X    = (const float*)d_in[1];
  const float* W1   = (const float*)d_in[2];
  const float* b1   = (const float*)d_in[3];
  const float* W2   = (const float*)d_in[4];
  const float* b2   = (const float*)d_in[5];
  const float* A    = (const float*)d_in[6];
  const float* Wmu  = (const float*)d_in[7];
  const float* bmu  = (const float*)d_in[8];
  const float* Wmu2 = (const float*)d_in[9];
  const float* bmu2 = (const float*)d_in[10];
  const float* Wsig = (const float*)d_in[11];
  const float* bsig = (const float*)d_in[12];
  const float* Wal  = (const float*)d_in[13];
  const float* bal  = (const float*)d_in[14];

  char* ws = (char*)d_ws;
  float*          H1  = (float*)ws;                                   // 131072 B
  unsigned short* Ebf = (unsigned short*)(ws + 131072);               //  65536 B
  unsigned short* Bws = (unsigned short*)(ws + 131072 + 65536);       // 33554432 B
  unsigned short* Ct  = (unsigned short*)(ws + 131072 + 65536 + 33554432);

  k_enc1<<<WIN, 128, 0, stream>>>(X, W1, b1, H1);
  k_enc2<<<WIN, 128, 0, stream>>>(H1, W2, b2, Ebf);
  k_core<<<dim3(RDIM, 8), 128, 0, stream>>>(A, Ebf, Bws);
  k_transpose<<<dim3(WIN, 8, 8), 256, 0, stream>>>(Bws, Ct);
  k_recur<<<1, 1024, 0, stream>>>(Ct, prev, X, Wmu, bmu, Wmu2, bmu2,
                                  Wsig, bsig, Wal, bal, (float*)d_out);
}
